// FlaGatedDeltaNetSequenceMixer_47064251630309
// MI455X (gfx1250) — matrix operand order repaired and checked by reading, not compiled, not run on hardware
//
#include <hip/hip_runtime.h>
#include <hip/hip_bf16.h>

// ---------------------------------------------------------------------------
// Types
// ---------------------------------------------------------------------------
typedef __attribute__((ext_vector_type(16))) __bf16 v16bf;
typedef __attribute__((ext_vector_type(8)))  __bf16 v8bf;
typedef __attribute__((ext_vector_type(8)))  float  v8f;
typedef __attribute__((ext_vector_type(4)))  unsigned int v4u;
typedef __attribute__((ext_vector_type(8)))  int v8i;
typedef __attribute__((ext_vector_type(4)))  int v4i;

union FragU { v16bf v; v8bf h[2]; };

#define LDS_STRIDE 40   // 32 bf16 payload + 8 bf16 pad = 80B rows (16B aligned, odd bank stride)

#if defined(__HIP_DEVICE_COMPILE__) && __has_builtin(__builtin_amdgcn_tensor_load_to_lds) && __has_builtin(__builtin_amdgcn_s_wait_tensorcnt)
#define HAS_TDM 1
#else
#define HAS_TDM 0
#endif

#if HAS_TDM
#define WAIT_TENSOR(n) __builtin_amdgcn_s_wait_tensorcnt(n)
#else
#define WAIT_TENSOR(n)
#endif

__device__ __forceinline__ unsigned short f32_to_bf16(float f) {
    union { float f; unsigned int u; } v; v.f = f;
    unsigned int u = v.u;
    u += 0x7FFFu + ((u >> 16) & 1u);   // round-to-nearest-even
    return (unsigned short)(u >> 16);
}

__device__ __forceinline__ float silu(float x) { return x / (1.f + expf(-x)); }

// ---------------------------------------------------------------------------
// TDM tile load: 2D tile, 32 bf16 per row (64B), tile_h rows, pad 16B per row
// so LDS row stride = 80B = LDS_STRIDE bf16.  D# layout per CDNA5 ISA ch.10.8.
// ---------------------------------------------------------------------------
#if HAS_TDM
__device__ __forceinline__ void tdm_load_tile(const void* gptr, unsigned lds_byte,
                                              unsigned tile_h, unsigned row_stride_elems) {
    unsigned long long ga = (unsigned long long)gptr;
    v4u g0;
    g0[0] = 1u;                                              // count=1, no gather
    g0[1] = lds_byte;                                        // lds_addr
    g0[2] = (unsigned)ga;                                    // global_addr[31:0]
    g0[3] = (unsigned)((ga >> 32) & 0x01FFFFFFu) | 0x80000000u; // ga[56:32] | type=2
    v8i g1;
    // data_size=2B (1<<16), pad_enable (1<<20), pad_interval=16dw (3<<22), pad_amount=4dw (3<<25)
    g1[0] = (int)((1u << 16) | (1u << 20) | (3u << 22) | (3u << 25));
    g1[1] = (int)(32u << 16);                                // tensor_dim0 = 32
    g1[2] = (int)((tile_h & 0xFFFFu) << 16);                 // tensor_dim1
    g1[3] = (int)(32u << 16);                                // tile_dim0 = 32
    g1[4] = (int)(tile_h & 0xFFFFu);                         // tile_dim1, tile_dim2=0
    g1[5] = (int)row_stride_elems;                           // tensor_dim0_stride[31:0]
    g1[6] = 0;
    g1[7] = 0;
    v4i z4 = {0, 0, 0, 0};
#if __clang_major__ >= 23
    v8i z8 = {0, 0, 0, 0, 0, 0, 0, 0};
    __builtin_amdgcn_tensor_load_to_lds(g0, g1, z4, z4, z8, 0);
#else
    __builtin_amdgcn_tensor_load_to_lds(g0, g1, z4, z4, 0);
#endif
}
#endif

// ---------------------------------------------------------------------------
// Fragment loads from LDS, contiguous k order k = 16*(lane/16) + s for both A 16x32 and B 32x16
// ---------------------------------------------------------------------------
__device__ __forceinline__ v16bf load_a_frag(const __bf16* lds, int row0, int lane) {
    int r  = lane & 15;
    int hi = (lane >> 4) ? 16 : 0;           // lanes 16..31 hold K 16..31
    const __bf16* p = lds + (row0 + r) * LDS_STRIDE;
    FragU f;
    f.h[0] = *(const v8bf*)(p + hi);         // K +0..7
    f.h[1] = *(const v8bf*)(p + 8 + hi);     // K +8..15
    return f.v;
}

__device__ __forceinline__ v16bf load_b_frag(const __bf16* lds, int col0, int lane) {
    int r  = lane & 15;
    int ko = (lane >> 4) ? 16 : 0;           // lanes 16..31 hold K 16..31
    const __bf16* p = lds + (col0 + r) * LDS_STRIDE + ko;
    FragU f;
    f.h[0] = *(const v8bf*)p;                // K +0..7
    f.h[1] = *(const v8bf*)(p + 8);          // K +8..15
    return f.v;
}

__device__ __forceinline__ v8f wmma_bf16(v16bf a, v16bf b, v8f c) {
    return __builtin_amdgcn_wmma_f32_16x16x32_bf16(false, a, false, b, (short)0, c, false, false);
}

// ---------------------------------------------------------------------------
// GEMM: C[M,N] (f32) = A[M,K] (bf16, row major) x W[N,K]^T (bf16, row major)
// block = 256 thr (8 waves), block tile 128x64, wave tile 32x32, K step 32
// ---------------------------------------------------------------------------
__device__ __forceinline__ void gemm_stage(const __bf16* A, const __bf16* W, int K,
                                           int m0, int n0, int kb,
                                           __bf16* bufA, __bf16* bufB, int tid) {
#if HAS_TDM
    if (tid < 32) {   // wave 0 issues both DMA descriptors
        tdm_load_tile(A + (size_t)m0 * K + kb,
                      (unsigned)(unsigned long long)(void*)bufA, 128u, (unsigned)K);
        tdm_load_tile(W + (size_t)n0 * K + kb,
                      (unsigned)(unsigned long long)(void*)bufB, 64u, (unsigned)K);
    }
#else
    #pragma unroll
    for (int j = 0; j < 2; ++j) {
        int c = tid + j * 256;               // 512 chunks of 8 bf16 for A
        int row = c >> 2, cc = (c & 3) << 3;
        *(v8bf*)(bufA + row * LDS_STRIDE + cc) =
            *(const v8bf*)(A + (size_t)(m0 + row) * K + kb + cc);
    }
    {
        int c = tid;                          // 256 chunks for B
        int row = c >> 2, cc = (c & 3) << 3;
        *(v8bf*)(bufB + row * LDS_STRIDE + cc) =
            *(const v8bf*)(W + (size_t)(n0 + row) * K + kb + cc);
    }
#endif
}

__global__ __launch_bounds__(256)
void gemm_bf16_wmma(const unsigned short* Ag, const unsigned short* Wg_,
                    float* C, int M, int N, int K) {
    __shared__ __align__(16) __bf16 ldsA[2][128 * LDS_STRIDE];
    __shared__ __align__(16) __bf16 ldsB[2][64 * LDS_STRIDE];
    const int tid  = (int)threadIdx.x;
    const int lane = tid & 31;
    const int wv   = tid >> 5;
    const int m0   = (int)blockIdx.y * 128;
    const int n0   = (int)blockIdx.x * 64;
    const int wm   = (wv & 3) * 32;
    const int wn   = (wv >> 2) * 32;
    const __bf16* A = (const __bf16*)Ag;
    const __bf16* W = (const __bf16*)Wg_;
    v8f c00 = {}, c01 = {}, c10 = {}, c11 = {};
    const int nk = K >> 5;

    gemm_stage(A, W, K, m0, n0, 0, ldsA[0], ldsB[0], tid);
    for (int kt = 0; kt < nk; ++kt) {
        const int cur = kt & 1, nxt = cur ^ 1;
        if (kt + 1 < nk) {
            gemm_stage(A, W, K, m0, n0, (kt + 1) << 5, ldsA[nxt], ldsB[nxt], tid);
            WAIT_TENSOR(2);                  // current buffer's 2 DMAs complete
        } else {
            WAIT_TENSOR(0);
        }
        __syncthreads();
        const __bf16* lA = ldsA[cur];
        const __bf16* lB = ldsB[cur];
        v16bf a0 = load_a_frag(lA, wm,      lane);
        v16bf a1 = load_a_frag(lA, wm + 16, lane);
        v16bf b0 = load_b_frag(lB, wn,      lane);
        v16bf b1 = load_b_frag(lB, wn + 16, lane);
        c00 = wmma_bf16(a0, b0, c00);
        c01 = wmma_bf16(a0, b1, c01);
        c10 = wmma_bf16(a1, b0, c10);
        c11 = wmma_bf16(a1, b1, c11);
        __syncthreads();
    }
    // C/D layout: lane<16 -> N=lane, vgpr r -> M=r ; lane>=16 -> M=r+8
    const int r  = lane & 15;
    const int ra = (lane >> 4) * 8;
    #pragma unroll
    for (int i = 0; i < 8; ++i) {
        C[(size_t)(m0 + wm + ra + i)      * N + (n0 + wn + r)]      = c00[i];
        C[(size_t)(m0 + wm + ra + i)      * N + (n0 + wn + 16 + r)] = c01[i];
        C[(size_t)(m0 + wm + 16 + ra + i) * N + (n0 + wn + r)]      = c10[i];
        C[(size_t)(m0 + wm + 16 + ra + i) * N + (n0 + wn + 16 + r)] = c11[i];
    }
}

// ---------------------------------------------------------------------------
// f32 -> bf16 convert
// ---------------------------------------------------------------------------
__global__ void cvt_bf16_kernel(const float* x, unsigned short* y, int n) {
    int i = (int)blockIdx.x * (int)blockDim.x + (int)threadIdx.x;
    if (i < n) y[i] = f32_to_bf16(x[i]);
}

// ---------------------------------------------------------------------------
// causal depthwise conv (K=4) + SiLU, in place. one thread per (b, channel)
// ---------------------------------------------------------------------------
__global__ void dwconv_silu_inplace(float* x, const float* w, int Bv, int Tv, int C) {
    int gid = (int)blockIdx.x * (int)blockDim.x + (int)threadIdx.x;
    if (gid >= Bv * C) return;
    int b = gid / C, c = gid % C;
    float w0 = w[c * 4 + 0], w1 = w[c * 4 + 1], w2 = w[c * 4 + 2], w3 = w[c * 4 + 3];
    float x0 = 0.f, x1 = 0.f, x2 = 0.f;
    float* p = x + (size_t)b * Tv * C + c;
    for (int t = 0; t < Tv; ++t) {
        float xt = p[(size_t)t * C];
        float y  = w0 * x0 + w1 * x1 + w2 * x2 + w3 * xt;
        p[(size_t)t * C] = silu(y);
        x0 = x1; x1 = x2; x2 = xt;
    }
}

// ---------------------------------------------------------------------------
// L2 norm over rows of 64 (one wave32 per row; 2 elems/lane)
// ---------------------------------------------------------------------------
__global__ void l2norm_rows(float* x, int nrows, float scale) {
    int w    = ((int)blockIdx.x * (int)blockDim.x + (int)threadIdx.x) >> 5;
    int lane = (int)threadIdx.x & 31;
    if (w >= nrows) return;
    float* p = x + (size_t)w * 64;
    float a = p[lane], b = p[lane + 32];
    float s = a * a + b * b;
    for (int off = 16; off; off >>= 1) s += __shfl_xor(s, off, 32);
    float r = rsqrtf(s + 1e-6f) * scale;
    p[lane] = a * r; p[lane + 32] = b * r;
}

// ---------------------------------------------------------------------------
// beta = sigmoid(h.Wb^T), g = -exp(A_log)*softplus(h.Wa^T + dt_bias)
// one wave per (token, head); two 1024-dot products
// ---------------------------------------------------------------------------
__global__ void betag_kernel(const float* hidden, const float* Wb, const float* Wa,
                             const float* A_log, const float* dt_bias,
                             float* beta, float* g, int ntok, int NHv, int K) {
    int w    = ((int)blockIdx.x * (int)blockDim.x + (int)threadIdx.x) >> 5;
    int lane = (int)threadIdx.x & 31;
    if (w >= ntok * NHv) return;
    int tok = w / NHv, h = w % NHv;
    const float* hp = hidden + (size_t)tok * K;
    const float* bp = Wb + (size_t)h * K;
    const float* ap = Wa + (size_t)h * K;
    float sb = 0.f, sa = 0.f;
    for (int i = lane; i < K; i += 32) { float x = hp[i]; sb += x * bp[i]; sa += x * ap[i]; }
    for (int off = 16; off; off >>= 1) {
        sb += __shfl_xor(sb, off, 32);
        sa += __shfl_xor(sa, off, 32);
    }
    if (lane == 0) {
        beta[w] = 1.f / (1.f + expf(-sb));
        float z  = sa + dt_bias[h];
        float sp = (z > 20.f) ? z : log1pf(expf(z));
        g[w] = -expf(A_log[h]) * sp;
    }
}

// ---------------------------------------------------------------------------
// delta-rule scan. one block per (b,h). 256 thr: col = tid&63, row-grp = tid>>6
// S[64x64] lives in registers: 16 rows of one column per thread.
// ---------------------------------------------------------------------------
__global__ __launch_bounds__(256)
void deltanet_scan(const float* q, const float* k, const float* v,
                   const float* g, const float* beta, float* o,
                   int Bv, int Tv, int NHv) {
    const int bh  = (int)blockIdx.x;
    const int b   = bh / NHv, h = bh % NHv;
    const int tid = (int)threadIdx.x;
    const int col = tid & 63;
    const int grp = tid >> 6;      // 0..3
    __shared__ float qs[64], ks[64], vs[64], dvs[64];
    __shared__ float red[4][64];
    __shared__ float sgb[2];
    float S[16];
    #pragma unroll
    for (int i = 0; i < 16; ++i) S[i] = 0.f;

    for (int t = 0; t < Tv; ++t) {
        const size_t base = (((size_t)b * Tv + t) * NHv + h) * 64;
        if (tid < 64)        qs[tid]       = q[base + tid];
        else if (tid < 128)  ks[tid - 64]  = k[base + tid - 64];
        else if (tid < 192)  vs[tid - 128] = v[base + tid - 128];
        else if (tid == 192) {
            size_t sb = ((size_t)b * Tv + t) * NHv + h;
            sgb[0] = expf(g[sb]);
            sgb[1] = beta[sb];
        }
        __syncthreads();
        const float dec = sgb[0];
        float p = 0.f;
        #pragma unroll
        for (int i = 0; i < 16; ++i) { S[i] *= dec; p += ks[grp * 16 + i] * S[i]; }
        red[grp][col] = p;
        __syncthreads();
        if (tid < 64) {
            float pr = red[0][tid] + red[1][tid] + red[2][tid] + red[3][tid];
            dvs[tid] = (vs[tid] - pr) * sgb[1];
        }
        __syncthreads();
        const float dvv = dvs[col];
        float op = 0.f;
        #pragma unroll
        for (int i = 0; i < 16; ++i) { S[i] += ks[grp * 16 + i] * dvv; op += qs[grp * 16 + i] * S[i]; }
        red[grp][col] = op;
        __syncthreads();
        if (tid < 64) o[base + tid] = red[0][tid] + red[1][tid] + red[2][tid] + red[3][tid];
        __syncthreads();
    }
}

// ---------------------------------------------------------------------------
// gated RMSNorm over rows of 64 + silu(gate), emit bf16 for final GEMM
// ---------------------------------------------------------------------------
__global__ void gated_rmsnorm_bf16(const float* o, const float* gate, const float* nw,
                                   unsigned short* out, int nrows) {
    int w    = ((int)blockIdx.x * (int)blockDim.x + (int)threadIdx.x) >> 5;
    int lane = (int)threadIdx.x & 31;
    if (w >= nrows) return;
    const float* op = o    + (size_t)w * 64;
    const float* gp = gate + (size_t)w * 64;
    float a = op[lane], b = op[lane + 32];
    float s = a * a + b * b;
    for (int off = 16; off; off >>= 1) s += __shfl_xor(s, off, 32);
    float r = rsqrtf(s * (1.f / 64.f) + 1e-5f);
    float y0 = a * r * nw[lane]      * silu(gp[lane]);
    float y1 = b * r * nw[lane + 32] * silu(gp[lane + 32]);
    out[(size_t)w * 64 + lane]      = f32_to_bf16(y0);
    out[(size_t)w * 64 + lane + 32] = f32_to_bf16(y1);
}

// ---------------------------------------------------------------------------
// launch
// ---------------------------------------------------------------------------
extern "C" void kernel_launch(void* const* d_in, const int* in_sizes, int n_in,
                              void* d_out, int out_size, void* d_ws, size_t ws_size,
                              hipStream_t stream) {
    const float* hidden  = (const float*)d_in[0];
    const float* Wq      = (const float*)d_in[1];
    const float* Wk      = (const float*)d_in[2];
    const float* Wv      = (const float*)d_in[3];
    const float* conv_q  = (const float*)d_in[4];
    const float* conv_k  = (const float*)d_in[5];
    const float* conv_v  = (const float*)d_in[6];
    const float* Wb      = (const float*)d_in[7];
    const float* Wa      = (const float*)d_in[8];
    const float* A_log   = (const float*)d_in[9];
    const float* dt_bias = (const float*)d_in[10];
    const float* Wg      = (const float*)d_in[11];
    const float* norm_w  = (const float*)d_in[12];
    const float* Wo      = (const float*)d_in[13];

    const int Bv = 2, Tv = 2048, Hv = 1024, NHv = 16;
    const int MT = Bv * Tv;               // 4096 tokens

    char* ws = (char*)d_ws;
    auto alloc = [&](size_t bytes) -> char* {
        char* p = ws; ws += (bytes + 255) & ~(size_t)255; return p;
    };
    unsigned short* Hbf  = (unsigned short*)alloc((size_t)MT * Hv * 2);
    unsigned short* Wqb  = (unsigned short*)alloc((size_t)Hv * Hv * 2);
    unsigned short* Wkb  = (unsigned short*)alloc((size_t)Hv * Hv * 2);
    unsigned short* Wvb  = (unsigned short*)alloc((size_t)Hv * Hv * 2);
    unsigned short* Wgb  = (unsigned short*)alloc((size_t)Hv * Hv * 2);
    unsigned short* Wob  = (unsigned short*)alloc((size_t)Hv * Hv * 2);
    float* qraw = (float*)alloc((size_t)MT * Hv * 4);
    float* kraw = (float*)alloc((size_t)MT * Hv * 4);
    float* vraw = (float*)alloc((size_t)MT * Hv * 4);
    float* graw = (float*)alloc((size_t)MT * Hv * 4);
    float* beta = (float*)alloc((size_t)MT * NHv * 4);
    float* gdec = (float*)alloc((size_t)MT * NHv * 4);
    float* obuf = (float*)alloc((size_t)MT * Hv * 4);
    unsigned short* Obf = (unsigned short*)alloc((size_t)MT * Hv * 2);

    // 1) bf16 conversion of activations + weights
    cvt_bf16_kernel<<<(MT * Hv + 255) / 256, 256, 0, stream>>>(hidden, Hbf, MT * Hv);
    cvt_bf16_kernel<<<(Hv * Hv + 255) / 256, 256, 0, stream>>>(Wq, Wqb, Hv * Hv);
    cvt_bf16_kernel<<<(Hv * Hv + 255) / 256, 256, 0, stream>>>(Wk, Wkb, Hv * Hv);
    cvt_bf16_kernel<<<(Hv * Hv + 255) / 256, 256, 0, stream>>>(Wv, Wvb, Hv * Hv);
    cvt_bf16_kernel<<<(Hv * Hv + 255) / 256, 256, 0, stream>>>(Wg, Wgb, Hv * Hv);
    cvt_bf16_kernel<<<(Hv * Hv + 255) / 256, 256, 0, stream>>>(Wo, Wob, Hv * Hv);

    // 2) WMMA projections: q/k/v/gate = hidden @ W^T
    dim3 ggrid(Hv / 64, MT / 128);
    gemm_bf16_wmma<<<ggrid, 256, 0, stream>>>(Hbf, Wqb, qraw, MT, Hv, Hv);
    gemm_bf16_wmma<<<ggrid, 256, 0, stream>>>(Hbf, Wkb, kraw, MT, Hv, Hv);
    gemm_bf16_wmma<<<ggrid, 256, 0, stream>>>(Hbf, Wvb, vraw, MT, Hv, Hv);
    gemm_bf16_wmma<<<ggrid, 256, 0, stream>>>(Hbf, Wgb, graw, MT, Hv, Hv);

    // 3) causal depthwise conv + SiLU (in place)
    int cgrid = (Bv * Hv + 255) / 256;
    dwconv_silu_inplace<<<cgrid, 256, 0, stream>>>(qraw, conv_q, Bv, Tv, Hv);
    dwconv_silu_inplace<<<cgrid, 256, 0, stream>>>(kraw, conv_k, Bv, Tv, Hv);
    dwconv_silu_inplace<<<cgrid, 256, 0, stream>>>(vraw, conv_v, Bv, Tv, Hv);

    // 4) L2 norm (q also scaled by d^-0.5)
    int nrows = MT * NHv;                  // 65536 rows of 64
    l2norm_rows<<<nrows / 8, 256, 0, stream>>>(qraw, nrows, 0.125f);
    l2norm_rows<<<nrows / 8, 256, 0, stream>>>(kraw, nrows, 1.0f);

    // 5) beta / gate-decay head projections
    betag_kernel<<<nrows / 8, 256, 0, stream>>>(hidden, Wb, Wa, A_log, dt_bias,
                                                beta, gdec, MT, NHv, Hv);

    // 6) sequential delta-rule scan
    deltanet_scan<<<Bv * NHv, 256, 0, stream>>>(qraw, kraw, vraw, gdec, beta, obuf,
                                                Bv, Tv, NHv);

    // 7) gated RMSNorm -> bf16
    gated_rmsnorm_bf16<<<nrows / 8, 256, 0, stream>>>(obuf, graw, norm_w, Obf, nrows);

    // 8) output projection (WMMA) -> f32 d_out
    gemm_bf16_wmma<<<ggrid, 256, 0, stream>>>(Obf, Wob, (float*)d_out, MT, Hv, Hv);
}
